// MultiHeadLatentAttention_64278480552158
// MI455X (gfx1250) — matrix operand order repaired and checked by reading, not compiled, not run on hardware
//
#include <hip/hip_runtime.h>
#include <hip/hip_bf16.h>

typedef __bf16    v16bf __attribute__((ext_vector_type(16)));
typedef float     v8f   __attribute__((ext_vector_type(8)));
typedef unsigned  v8u   __attribute__((ext_vector_type(8)));

__device__ __forceinline__ __bf16 f2bf(float f) {
    unsigned u = __builtin_bit_cast(unsigned, f);
    u += 0x7FFFu + ((u >> 16) & 1u);          // round-to-nearest-even
    unsigned short h = (unsigned short)(u >> 16);
    return __builtin_bit_cast(__bf16, h);
}

__device__ __forceinline__ v8f wmma_bf16(v16bf a, v16bf b, v8f c) {
    return __builtin_amdgcn_wmma_f32_16x16x32_bf16(
        /*neg_a=*/false, a, /*neg_b=*/false, b,
        /*c_mod=*/(short)0, c, /*reuse_a=*/false, /*reuse_b=*/false);
}

// K-offset (within a 32-wide chunk) for the 16-bit A-matrix VGPR layout:
// lanes 0-15: VGPR0-3 -> K 0..7, VGPR4-7 -> K 16..23; lanes 16-31: +8.
__device__ __forceinline__ int a_k_off(int j, int kgrp) {
    return kgrp * 8 + (j < 4 ? 2 * j : 16 + 2 * (j - 4));
}

// CDNA5 async global->LDS copy (16 bytes / lane), tracked by ASYNCcnt.
// LDS address = low 32 bits of the generic pointer.
__device__ __forceinline__ void async_copy_b128(void* lds_dst, const void* gsrc) {
    unsigned l = (unsigned)(uintptr_t)lds_dst;
    asm volatile("global_load_async_to_lds_b128 %0, %1, off"
                 :: "v"(l), "v"(gsrc) : "memory");
}

// ---------------------------------------------------------------------------
// Tiled bf16 GEMM: C(MxN) = A(MxK) @ B(KxN), row-major, f32 accumulate.
// Block tile 128xBNT, K tile 64, 256 threads = 8 waves (4x2),
// wave tile 32 x (BNT/2). A tile double-buffered via async-to-LDS copies.
// OUTF=1 -> fp32 C, OUTF=0 -> bf16 C.
// ---------------------------------------------------------------------------
template <int OUTF, int BNT>
__global__ __launch_bounds__(256) void gemm_bf16_kernel(
    const __bf16* __restrict__ A, const __bf16* __restrict__ Bm,
    float* __restrict__ Cf, __bf16* __restrict__ Cb, int M, int N, int K) {
    constexpr int BM = 128, BK = 64;
    constexpr int NSUB = BNT / 32;                 // 16-wide n-subtiles per wave
    __shared__ alignas(16) __bf16 As[2][BM * BK];  // [buf][row][k]
    __shared__ alignas(16) __bf16 Bs[BNT * BK];    // transposed: [n][k]
    unsigned* Bs_u = (unsigned*)Bs;

    const int tid  = threadIdx.x;
    const int lane = tid & 31;
    const int wave = tid >> 5;
    const int wm   = wave >> 1;                    // 0..3 -> 32 rows each
    const int wn   = wave & 1;                     // 0..1 -> BNT/2 cols each
    const int m0   = blockIdx.y * BM;
    const int n0   = blockIdx.x * BNT;

    const int ml   = lane & 15;
    const int kgrp = lane >> 4;

    v8f acc[2][NSUB];
#pragma unroll
    for (int i = 0; i < 2; ++i)
#pragma unroll
        for (int j = 0; j < NSUB; ++j) acc[i][j] = {};

    // async staging of one A tile: 1024 x 16B chunks, 4 per thread
    auto stageA = [&](int buf, int k0) {
#pragma unroll
        for (int i = 0; i < 4; ++i) {
            int c2  = tid + i * 256;               // 0..1023
            int r   = c2 >> 3;                     // 8 chunks per 64-elem row
            int c16 = c2 & 7;
            async_copy_b128(&As[buf][r * BK + c16 * 8],
                            A + (size_t)(m0 + r) * K + k0 + c16 * 8);
        }
    };

    stageA(0, 0);
    const int ntile = K / BK;
    for (int t = 0; t < ntile; ++t) {
        const int k0  = t * BK;
        const int cur = t & 1;
        if (t + 1 < ntile) stageA(cur ^ 1, k0 + BK);   // overlap next A copy

        // stage B tile transposed so per-lane frag reads are contiguous in k
        for (int q = tid; q < BK * BNT; q += 256) {
            int kk = q / BNT;
            int nn = q % BNT;
            Bs[nn * BK + kk] = Bm[(size_t)(k0 + kk) * N + n0 + nn];
        }

        // current tile's 4 async copies done (in-order completion)
        if (t + 1 < ntile) asm volatile("s_wait_asynccnt 4" ::: "memory");
        else               asm volatile("s_wait_asynccnt 0" ::: "memory");
        __syncthreads();

        unsigned* As_u = (unsigned*)As[cur];
#pragma unroll
        for (int kk = 0; kk < BK; kk += 32) {
            v16bf afr[2], bfr[NSUB];
#pragma unroll
            for (int i = 0; i < 2; ++i) {
                v8u au;
#pragma unroll
                for (int j = 0; j < 8; ++j) {
                    int k = kk + a_k_off(j, kgrp);
                    au[j] = As_u[((wm * 32 + i * 16 + ml) * BK + k) >> 1];
                }
                afr[i] = __builtin_bit_cast(v16bf, au);
            }
#pragma unroll
            for (int j2 = 0; j2 < NSUB; ++j2) {
                v8u bu;
                int nb = wn * (BNT / 2) + j2 * 16 + ml;
                int kb = kk;
#pragma unroll
                for (int j = 0; j < 8; ++j)
                    bu[j] = Bs_u[(nb * BK + kb + a_k_off(j, kgrp)) >> 1];
                bfr[j2] = __builtin_bit_cast(v16bf, bu);
            }
#pragma unroll
            for (int i = 0; i < 2; ++i)
#pragma unroll
                for (int j2 = 0; j2 < NSUB; ++j2)
                    acc[i][j2] = wmma_bf16(afr[i], bfr[j2], acc[i][j2]);
        }
        __syncthreads();
    }

    // epilogue: C layout — lane = N, VGPR r = M row (lanes 16-31: +8)
    const int g = lane >> 4;
#pragma unroll
    for (int i = 0; i < 2; ++i)
#pragma unroll
        for (int j2 = 0; j2 < NSUB; ++j2)
#pragma unroll
            for (int r = 0; r < 8; ++r) {
                int mm = m0 + wm * 32 + i * 16 + r + 8 * g;
                int nc = n0 + wn * (BNT / 2) + j2 * 16 + ml;
                float v = acc[i][j2][r];
                if (OUTF) Cf[(size_t)mm * N + nc] = v;
                else      Cb[(size_t)mm * N + nc] = f2bf(v);
            }
}

// ---------------------------------------------------------------------------
// Flash attention (causal), bf16 WMMA, online softmax.
// 128 threads = 4 waves; each wave owns 16 query rows. grid = (S/64, B*H).
// q,k layout (B,H,S,128); v layout (B,H,128,S) transposed; out (B,S,H,128).
// ---------------------------------------------------------------------------
__global__ __launch_bounds__(128) void mla_attn_kernel(
    const __bf16* __restrict__ qg, const __bf16* __restrict__ kg,
    const __bf16* __restrict__ vg, __bf16* __restrict__ og,
    int Bn, int Hn, int Sn) {
    __shared__ __bf16 Pl[4 * 16 * 32];          // per-wave exp(P) staging tile
    unsigned* Plu = (unsigned*)Pl;

    const int lane  = threadIdx.x & 31;
    const int w     = threadIdx.x >> 5;
    const int bh    = blockIdx.y;
    const int b     = bh / Hn;
    const int h     = bh % Hn;
    const int qbase = blockIdx.x * 64 + w * 16;

    const __bf16* qp = qg + (size_t)bh * Sn * 128;
    const __bf16* kp = kg + (size_t)bh * Sn * 128;
    const __bf16* vp = vg + (size_t)bh * 128 * Sn;

    const int ml   = lane & 15;
    const int g    = lane >> 4;
    const int kg8  = g * 8;
    const int kg16 = g * 16;

    // Q fragments: 16x128 held as 4 A-frags, resident for the whole kernel
    v16bf qa[4];
#pragma unroll
    for (int c = 0; c < 4; ++c) {
        v8u au;
#pragma unroll
        for (int j = 0; j < 8; ++j) {
            int k = c * 32 + kg8 + (j < 4 ? 2 * j : 16 + 2 * (j - 4));
            au[j] = *(const unsigned*)(qp + (size_t)(qbase + ml) * 128 + k);
        }
        qa[c] = __builtin_bit_cast(v16bf, au);
    }

    v8f O[8] = {};
    float mi[8], li[8];
#pragma unroll
    for (int r = 0; r < 8; ++r) { mi[r] = -3.0e38f; li[r] = 0.0f; }

    const float sc = 0.08838834764831845f;      // 1/sqrt(128)

    for (int kv0 = 0; kv0 < qbase + 16; kv0 += 32) {
        // ---- scores: Q(16x128) @ K^T -> two 16x16 tiles (kv cols 0-15,16-31)
        v8f s0 = {}, s1 = {};
#pragma unroll
        for (int c = 0; c < 4; ++c) {
            v8u b0u, b1u;
#pragma unroll
            for (int j = 0; j < 8; ++j) {
                int hd = c * 32 + kg8 + (j < 4 ? 2 * j : 16 + 2 * (j - 4));
                b0u[j] = *(const unsigned*)(kp + (size_t)(kv0 + ml) * 128 + hd);
                b1u[j] = *(const unsigned*)(kp + (size_t)(kv0 + 16 + ml) * 128 + hd);
            }
            s0 = wmma_bf16(qa[c], __builtin_bit_cast(v16bf, b0u), s0);
            s1 = wmma_bf16(qa[c], __builtin_bit_cast(v16bf, b1u), s1);
        }

        // ---- causal mask + online softmax (row = VGPR index, cols = lanes)
#pragma unroll
        for (int r = 0; r < 8; ++r) {
            int qrow = qbase + r + 8 * g;
            int c0 = kv0 + ml, c1 = c0 + 16;
            float v0 = (c0 <= qrow) ? s0[r] * sc : -3.0e38f;
            float v1 = (c1 <= qrow) ? s1[r] * sc : -3.0e38f;
            float rm = fmaxf(v0, v1);
#pragma unroll
            for (int off = 1; off < 16; off <<= 1)
                rm = fmaxf(rm, __shfl_xor(rm, off, 16));
            float mnew  = fmaxf(mi[r], rm);
            float alpha = __expf(mi[r] - mnew);
            float p0 = __expf(v0 - mnew);
            float p1 = __expf(v1 - mnew);
            float rs = p0 + p1;
#pragma unroll
            for (int off = 1; off < 16; off <<= 1)
                rs += __shfl_xor(rs, off, 16);
            li[r] = li[r] * alpha + rs;
            mi[r] = mnew;
#pragma unroll
            for (int nt = 0; nt < 8; ++nt) O[nt][r] *= alpha;
            int mrow = r + 8 * g;
            Pl[w * 512 + mrow * 32 + ml]      = f2bf(p0);
            Pl[w * 512 + mrow * 32 + 16 + ml] = f2bf(p1);
        }

        // ---- re-stage exp(P) (16x32) from LDS into A-fragment layout
        v8u pu;
#pragma unroll
        for (int j = 0; j < 8; ++j) {
            int k = kg8 + (j < 4 ? 2 * j : 16 + 2 * (j - 4));
            pu[j] = Plu[w * 256 + ((ml * 32 + k) >> 1)];
        }
        v16bf pa = __builtin_bit_cast(v16bf, pu);

        // ---- O += P @ V (V pre-transposed: contiguous along kv per lane)
#pragma unroll
        for (int nt = 0; nt < 8; ++nt) {
            v8u vu;
#pragma unroll
            for (int j = 0; j < 8; ++j) {
                int kvr = kv0 + kg8 + (j < 4 ? 2 * j : 16 + 2 * (j - 4));
                vu[j] = *(const unsigned*)(vp + (size_t)(nt * 16 + ml) * Sn + kvr);
            }
            O[nt] = wmma_bf16(pa, __builtin_bit_cast(v16bf, vu), O[nt]);
        }
    }

    // ---- normalize & store to (B,S,H,128)
#pragma unroll
    for (int r = 0; r < 8; ++r) {
        float inv = 1.0f / li[r];
        int srow = qbase + r + 8 * g;
        size_t base = (((size_t)b * Sn + srow) * Hn + h) * 128;
#pragma unroll
        for (int nt = 0; nt < 8; ++nt)
            og[base + nt * 16 + ml] = f2bf(O[nt][r] * inv);
    }
}

// ---------------------------------------------------------------------------
// Elementwise helpers
// ---------------------------------------------------------------------------
__global__ void convert_f32_bf16(const float* __restrict__ in,
                                 __bf16* __restrict__ out, size_t n) {
    size_t i = (size_t)blockIdx.x * blockDim.x + threadIdx.x;
    if (i < n) out[i] = f2bf(in[i]);
}

// q: (B,S,H,128) fp32 -> RoPE first 64 dims -> (B,H,S,128) bf16
__global__ void rope_q_kernel(const float* __restrict__ qf,
                              __bf16* __restrict__ qb, int Bn, int Hn, int Sn) {
    size_t idx = (size_t)blockIdx.x * blockDim.x + threadIdx.x;
    size_t total = (size_t)Bn * Hn * Sn * 128;
    if (idx >= total) return;
    int d = (int)(idx & 127);
    size_t t = idx >> 7;
    int s = (int)(t % Sn); t /= Sn;
    int h = (int)(t % Hn);
    int b = (int)(t / Hn);
    const float* row = qf + (((size_t)b * Sn + s) * Hn + h) * 128;
    float val;
    if (d < 64) {
        int j = d & 31;
        float freq = __expf(-(float)j * (9.210340371976184f / 32.0f)); // 10000^{-j/32}
        float e = (float)s * freq;
        float x = row[d];
        float rot = (d < 32) ? -row[d + 32] : row[d - 32];
        val = x * cosf(e) + rot * sinf(e);
    } else {
        val = row[d];
    }
    qb[idx] = f2bf(val);
}

// k = concat(rope(k_pe) broadcast over heads, k_unpe) -> (B,H,S,128) bf16
__global__ void assemble_k_kernel(const float* __restrict__ kunpe,
                                  const float* __restrict__ kpe,
                                  __bf16* __restrict__ kb, int Bn, int Hn, int Sn) {
    size_t idx = (size_t)blockIdx.x * blockDim.x + threadIdx.x;
    size_t total = (size_t)Bn * Hn * Sn * 128;
    if (idx >= total) return;
    int d = (int)(idx & 127);
    size_t t = idx >> 7;
    int s = (int)(t % Sn); t /= Sn;
    int h = (int)(t % Hn);
    int b = (int)(t / Hn);
    float val;
    if (d < 64) {
        const float* row = kpe + ((size_t)b * Sn + s) * 64;
        int j = d & 31;
        float freq = __expf(-(float)j * (9.210340371976184f / 32.0f));
        float e = (float)s * freq;
        float x = row[d];
        float rot = (d < 32) ? -row[d + 32] : row[d - 32];
        val = x * cosf(e) + rot * sinf(e);
    } else {
        val = kunpe[(((size_t)b * Sn + s) * Hn + h) * 64 + (d - 64)];
    }
    kb[idx] = f2bf(val);
}

// v: (B,S,H,128) fp32 -> (B,H,128,S) bf16 (transposed for P@V B-fragments)
__global__ void transpose_v_kernel(const float* __restrict__ vf,
                                   __bf16* __restrict__ vb, int Bn, int Hn, int Sn) {
    size_t idx = (size_t)blockIdx.x * blockDim.x + threadIdx.x;
    size_t total = (size_t)Bn * Hn * 128 * Sn;
    if (idx >= total) return;
    int s = (int)(idx % Sn);
    size_t t = idx / Sn;
    int hd = (int)(t % 128); t /= 128;
    int h  = (int)(t % Hn);
    int b  = (int)(t / Hn);
    vb[idx] = f2bf(vf[(((size_t)b * Sn + s) * Hn + h) * 128 + hd]);
}

// ---------------------------------------------------------------------------
extern "C" void kernel_launch(void* const* d_in, const int* in_sizes, int n_in,
                              void* d_out, int out_size, void* d_ws, size_t ws_size,
                              hipStream_t stream) {
    constexpr int Bn = 2, Sn = 2048, Dn = 4096, Hn = 32, DLn = 512;
    const float* x     = (const float*)d_in[0];
    const float* Wq    = (const float*)d_in[1];
    const float* Wdown = (const float*)d_in[2];
    const float* Wv    = (const float*)d_in[3];
    const float* Wk    = (const float*)d_in[4];
    const float* Wkpe  = (const float*)d_in[5];
    const float* Wo    = (const float*)d_in[6];
    // d_in[7] = mask: causal handled analytically in the attention kernel
    float* out = (float*)d_out;

    const size_t NX  = (size_t)Bn * Sn * Dn;   // 16.8M
    const size_t NWq = (size_t)Dn * Dn;
    const size_t NWd = (size_t)Dn * DLn;
    const size_t NWv = (size_t)DLn * Dn;
    const size_t NWk = (size_t)DLn * 2048;
    const size_t NWp = (size_t)DLn * 64;
    const size_t NWo = (size_t)Dn * Dn;
    const size_t NL  = (size_t)Bn * Sn * DLn;
    const size_t NKU = (size_t)Bn * Sn * 2048;
    const size_t NKP = (size_t)Bn * Sn * 64;

    char* ws = (char*)d_ws;
    size_t off = 0;
    auto alloc = [&](size_t bytes) -> void* {
        void* p = ws + off;
        off = (off + bytes + 255) & ~(size_t)255;
        return p;
    };

    __bf16* xb   = (__bf16*)alloc(NX * 2);
    __bf16* Wqb  = (__bf16*)alloc(NWq * 2);
    __bf16* Wdb  = (__bf16*)alloc(NWd * 2);
    __bf16* Wvb  = (__bf16*)alloc(NWv * 2);
    __bf16* Wkb  = (__bf16*)alloc(NWk * 2);
    __bf16* Wpb  = (__bf16*)alloc(NWp * 2);
    __bf16* Wob  = (__bf16*)alloc(NWo * 2);
    __bf16* latb = (__bf16*)alloc(NL * 2);
    __bf16* qb   = (__bf16*)alloc(NX * 2);
    __bf16* kb   = (__bf16*)alloc(NX * 2);
    __bf16* vbT  = (__bf16*)alloc(NX * 2);
    __bf16* attb = (__bf16*)alloc(NX * 2);
    float*  qf   = (float*)alloc(NX * 4);    // reused as v_f32 after RoPE consumes it
    float*  kuf  = (float*)alloc(NKU * 4);
    float*  kpf  = (float*)alloc(NKP * 4);

    auto nb = [](size_t n) { return (unsigned)((n + 255) / 256); };

    // fp32 -> bf16 staging
    convert_f32_bf16<<<nb(NX),  256, 0, stream>>>(x,     xb,  NX);
    convert_f32_bf16<<<nb(NWq), 256, 0, stream>>>(Wq,    Wqb, NWq);
    convert_f32_bf16<<<nb(NWd), 256, 0, stream>>>(Wdown, Wdb, NWd);
    convert_f32_bf16<<<nb(NWv), 256, 0, stream>>>(Wv,    Wvb, NWv);
    convert_f32_bf16<<<nb(NWk), 256, 0, stream>>>(Wk,    Wkb, NWk);
    convert_f32_bf16<<<nb(NWp), 256, 0, stream>>>(Wkpe,  Wpb, NWp);
    convert_f32_bf16<<<nb(NWo), 256, 0, stream>>>(Wo,    Wob, NWo);

    const int MR = Bn * Sn;   // 4096 rows

    // q = x @ Wq  -> RoPE -> (B,H,S,128) bf16
    gemm_bf16_kernel<1, 128><<<dim3(Dn / 128, MR / 128), 256, 0, stream>>>(
        xb, Wqb, qf, nullptr, MR, Dn, Dn);
    rope_q_kernel<<<nb(NX), 256, 0, stream>>>(qf, qb, Bn, Hn, Sn);

    // latent = x @ Wdown (bf16 out)
    gemm_bf16_kernel<0, 128><<<dim3(DLn / 128, MR / 128), 256, 0, stream>>>(
        xb, Wdb, nullptr, latb, MR, DLn, Dn);

    // v, k_unpe, k_pe from latent
    float* vf = qf;  // qf fully consumed by rope_q_kernel (same stream order)
    gemm_bf16_kernel<1, 128><<<dim3(Dn / 128, MR / 128), 256, 0, stream>>>(
        latb, Wvb, vf, nullptr, MR, Dn, DLn);
    gemm_bf16_kernel<1, 128><<<dim3(2048 / 128, MR / 128), 256, 0, stream>>>(
        latb, Wkb, kuf, nullptr, MR, 2048, DLn);
    gemm_bf16_kernel<1, 64><<<dim3(64 / 64, MR / 128), 256, 0, stream>>>(
        latb, Wpb, kpf, nullptr, MR, 64, DLn);

    assemble_k_kernel<<<nb(NX), 256, 0, stream>>>(kuf, kpf, kb, Bn, Hn, Sn);
    transpose_v_kernel<<<nb(NX), 256, 0, stream>>>(vf, vbT, Bn, Hn, Sn);

    // flash attention -> (B,S,H,128) bf16 == (B*S, D) for the final GEMM
    mla_attn_kernel<<<dim3(Sn / 64, Bn * Hn), 128, 0, stream>>>(
        qb, kb, vbT, attb, Bn, Hn, Sn);

    // out = attn @ Wo (fp32 out)
    gemm_bf16_kernel<1, 128><<<dim3(Dn / 128, MR / 128), 256, 0, stream>>>(
        attb, Wob, out, nullptr, MR, Dn, Dn);
}
